// lstm_4157528343251
// MI455X (gfx1250) — compile-verified, matrix operand order checked by reading, not run on hardware
//
#include <hip/hip_runtime.h>
#include <hip/hip_bf16.h>
#include <stdint.h>

// ---------------------------------------------------------------------------
// Types
// ---------------------------------------------------------------------------
typedef __bf16 bf16_t;
typedef __attribute__((ext_vector_type(16))) __bf16 v16bf;
typedef __attribute__((ext_vector_type(8)))  __bf16 v8bf;
typedef __attribute__((ext_vector_type(8)))  float  v8f;

// ---------------------------------------------------------------------------
// Problem constants
// ---------------------------------------------------------------------------
constexpr int kB  = 256;   // batch
constexpr int kT  = 1024;  // timesteps
constexpr int kD  = 64;    // input features (layer 0)
constexpr int kH  = 256;   // hidden
constexpr int kG4 = 4 * kH;  // 1024 gate columns

constexpr int NWG  = 64;   // persistent workgroups (must all be resident)
constexpr int NTHR = 256;  // 8 waves (wave32)
constexpr int LDS_PAD = 8; // bf16 elems of row padding -> conflict-free ds_load_b128
constexpr int CPT = (kB * kH) / (NWG * NTHR);  // cell elems per thread (= 4)

// ---------------------------------------------------------------------------
// Async global -> LDS copy (CDNA5), with fallback to plain copy
// ---------------------------------------------------------------------------
#if defined(__has_builtin)
#if __has_builtin(__builtin_amdgcn_global_load_async_to_lds_b128)
#define ASYNC_LDS 1
#endif
#endif
#ifndef ASYNC_LDS
#define ASYNC_LDS 0
#endif

// Builtin parameter type (verified from hipcc diagnostic in an earlier round):
//   '__attribute__((__vector_size__(4 * sizeof(int)))) int __device__ *'
typedef int v4i_vs __attribute__((vector_size(16)));
typedef __attribute__((address_space(1))) v4i_vs gv4i;  // prints as __device__
typedef __attribute__((address_space(3))) v4i_vs lv4i;  // LDS

__device__ __forceinline__ void copy16_g2l(void* lds_dst, const void* gsrc) {
#if ASYNC_LDS
  __builtin_amdgcn_global_load_async_to_lds_b128(
      (gv4i*)(unsigned long long)(size_t)gsrc,
      (lv4i*)(unsigned int)(size_t)lds_dst,
      0, 0);
#else
  *(uint4*)lds_dst = *(const uint4*)gsrc;
#endif
}

__device__ __forceinline__ void wait_lds_fill() {
#if ASYNC_LDS
#if __has_builtin(__builtin_amdgcn_s_wait_asynccnt)
  __builtin_amdgcn_s_wait_asynccnt(0);
#else
  asm volatile("s_wait_asynccnt 0" ::: "memory");
#endif
#endif
}

// ---------------------------------------------------------------------------
// WMMA helpers
// ---------------------------------------------------------------------------
__device__ __forceinline__ v8f wmma_bf16(v16bf a, v16bf b, v8f c) {
  // v_wmma_f32_16x16x32_bf16, D = A*B + C, f32 accumulate
  return __builtin_amdgcn_wmma_f32_16x16x32_bf16(
      /*neg_a=*/false, a, /*neg_b=*/false, b,
      /*c_mod=*/(short)0, c, /*reuse_a=*/false, /*reuse_b=*/false);
}

// bf16 16x32 A-frag / 32x16 B-frag loader per ISA 7.12.2:
// lane L: row = L&15, khalf = L>>4; vector elems 0..7 <- K = kb + khalf*8 + 0..7
//                                   vector elems 8..15 <- K = kb + 16 + khalf*8 + 0..7
__device__ __forceinline__ v16bf load_frag(const bf16_t* base, int stride,
                                           int lane, int kb) {
  const int r  = lane & 15;
  const int kh = lane >> 4;
  const bf16_t* p = base + (size_t)r * (size_t)stride + kb + kh * 8;
  v8bf lo = *(const v8bf*)(p);
  v8bf hi = *(const v8bf*)(p + 16);
  return __builtin_shufflevector(lo, hi, 0, 1, 2, 3, 4, 5, 6, 7,
                                         8, 9, 10, 11, 12, 13, 14, 15);
}

// ---------------------------------------------------------------------------
// Software grid barrier (all NWG workgroups must be resident)
// ---------------------------------------------------------------------------
__device__ __forceinline__ void grid_barrier(unsigned* bar) {
  __syncthreads();
  if (threadIdx.x == 0) {
    __threadfence();
    volatile unsigned* vb = (volatile unsigned*)bar;
    unsigned gen = vb[1];
    if (atomicAdd(&bar[0], 1u) == (unsigned)(NWG - 1)) {
      vb[0] = 0;
      __threadfence();
      vb[1] = gen + 1;
    } else {
      while (vb[1] == gen) { __builtin_amdgcn_s_sleep(1); }
    }
    __threadfence();
  }
  __syncthreads();
}

__device__ __forceinline__ float sigmoidf_(float x) {
  return 1.0f / (1.0f + __expf(-x));
}

// ---------------------------------------------------------------------------
// Small utility kernels
// ---------------------------------------------------------------------------
__global__ void cvt_kernel(bf16_t* __restrict__ dst, const float* __restrict__ src,
                           int n) {
  for (int i = blockIdx.x * blockDim.x + threadIdx.x; i < n;
       i += gridDim.x * blockDim.x)
    dst[i] = (bf16_t)src[i];
}

__global__ void bias_kernel(float* __restrict__ dst, const float* __restrict__ a,
                            const float* __restrict__ b, int n) {
  for (int i = blockIdx.x * blockDim.x + threadIdx.x; i < n;
       i += gridDim.x * blockDim.x)
    dst[i] = a[i] + b[i];
}

__global__ void zero_state_kernel(bf16_t* __restrict__ hb,
                                  unsigned* __restrict__ bar) {
  int i = blockIdx.x * blockDim.x + threadIdx.x;
  if (i == 0) { bar[0] = 0u; bar[1] = 0u; }
  for (; i < kB * kH; i += gridDim.x * blockDim.x) hb[i] = (bf16_t)0.0f;
}

// ---------------------------------------------------------------------------
// Persistent LSTM layer kernel.
//   Grid: 64 WGs x 256 threads. WG g: batch rows [(g>>4)*64, +64), gate cols
//   [(g&15)*64, +64). Each wave owns two 16x16 output tiles sharing the A
//   fragment. Recurrent weights (Whh slice) are pinned in VGPRs for the whole
//   1024-step scan; all A-fragments of a timestep are issued before any WMMA
//   so only one L2 latency is exposed per step. Cell state c lives in CPT=4
//   registers per thread across the entire scan (no memory round-trip).
//   For DIN==256 the Wih slice stays in LDS (async-filled once).
// ---------------------------------------------------------------------------
template <int DIN, bool IH_REG>
__global__ __launch_bounds__(NTHR) void lstm_layer_kernel(
    const bf16_t* __restrict__ xin,  // [B, T, DIN] bf16 (layer input sequence)
    const bf16_t* __restrict__ wih,  // [4H, DIN] bf16
    const bf16_t* __restrict__ whh,  // [4H, H] bf16
    const float* __restrict__ bias,  // [4H] = bih + bhh
    float* __restrict__ gates,       // [B, 4H] f32 scratch
    float* __restrict__ hfp,         // [B, H] f32 hidden (written at t = T-1)
    bf16_t* __restrict__ hbp,        // [B, H] bf16 hidden (GEMM A source)
    bf16_t* __restrict__ hs_out,     // [B, T, H] bf16 or nullptr
    unsigned* __restrict__ bar) {
  extern __shared__ bf16_t smem[];

  constexpr int KI  = DIN / 32;   // K-steps, input part
  constexpr int KR  = kH / 32;    // K-steps, recurrent part
  constexpr int ihrs = DIN + LDS_PAD;

  const int g    = blockIdx.x;
  const int mgrp = g >> 4;   // 0..3  -> 64 batch rows
  const int ngrp = g & 15;   // 0..15 -> 64 gate cols
  const int lane = threadIdx.x & 31;
  const int wave = threadIdx.x >> 5;
  const int ncol0 = ngrp * 64;

  // ---- tile assignment: 2 tiles/wave sharing the A fragment ---------------
  const int tid0 = wave * 2;
  const int mt   = tid0 >> 2;          // same for both tiles
  const int nt0  = tid0 & 3;
  const int nt1  = nt0 + 1;
  const int mbase = mgrp * 64 + mt * 16;

  // ---- Wih slice handling --------------------------------------------------
  const bf16_t* s_b0_ih = smem + (size_t)(nt0 * 16) * ihrs;
  const bf16_t* s_b1_ih = smem + (size_t)(nt1 * 16) * ihrs;
  if constexpr (!IH_REG) {
    // one-time async fill of the 64-row Wih slice into LDS (16B chunks)
    constexpr int CPR = DIN / 8;  // chunks per row
    for (int idx = threadIdx.x; idx < 64 * CPR; idx += NTHR) {
      int r = idx / CPR, kc = (idx - r * CPR) << 3;
      copy16_g2l(&smem[(size_t)r * ihrs + kc],
                 &wih[(size_t)(ncol0 + r) * DIN + kc]);
    }
    wait_lds_fill();
    __syncthreads();
  }

  // ---- loop-invariant weight fragments pinned in VGPRs ---------------------
  v16bf whr0[KR], whr1[KR];
  {
    const bf16_t* b0 = whh + (size_t)(ncol0 + nt0 * 16) * kH;
    const bf16_t* b1 = whh + (size_t)(ncol0 + nt1 * 16) * kH;
#pragma unroll
    for (int ki = 0; ki < KR; ++ki) {
      whr0[ki] = load_frag(b0, kH, lane, ki * 32);
      whr1[ki] = load_frag(b1, kH, lane, ki * 32);
    }
  }
  v16bf wir0[IH_REG ? KI : 1], wir1[IH_REG ? KI : 1];
  if constexpr (IH_REG) {
    const bf16_t* b0 = wih + (size_t)(ncol0 + nt0 * 16) * DIN;
    const bf16_t* b1 = wih + (size_t)(ncol0 + nt1 * 16) * DIN;
#pragma unroll
    for (int ki = 0; ki < KI; ++ki) {
      wir0[ki] = load_frag(b0, DIN, lane, ki * 32);
      wir1[ki] = load_frag(b1, DIN, lane, ki * 32);
    }
  }

  const bf16_t* Ah = hbp + (size_t)mbase * kH;

  // ---- per-thread resident cell state (CPT fixed (b,j) elements) ----------
  float creg[CPT];
#pragma unroll
  for (int k = 0; k < CPT; ++k) creg[k] = 0.0f;
  const int i0 = blockIdx.x * NTHR + threadIdx.x;

  for (int t = 0; t < kT; ++t) {
    // ============ phase 1: gates = [x_t | h_{t-1}] * [Wih|Whh]^T ===========
    v8f acc0 = {0, 0, 0, 0, 0, 0, 0, 0};
    v8f acc1 = {0, 0, 0, 0, 0, 0, 0, 0};

    // Issue ALL A-fragment loads up front (independent b128 loads -> the
    // scheduler can clause them and stagger s_wait_loadcnt per consumer).
    v16bf af[KI + KR];
    const bf16_t* Ain = xin + (size_t)mbase * (kT * DIN) + (size_t)t * DIN;
#pragma unroll
    for (int ki = 0; ki < KI; ++ki)
      af[ki] = load_frag(Ain, kT * DIN, lane, ki * 32);
#pragma unroll
    for (int ki = 0; ki < KR; ++ki)
      af[KI + ki] = load_frag(Ah, kH, lane, ki * 32);

#pragma unroll
    for (int ki = 0; ki < KI; ++ki) {
      if constexpr (IH_REG) {
        acc0 = wmma_bf16(af[ki], wir0[ki], acc0);
        acc1 = wmma_bf16(af[ki], wir1[ki], acc1);
      } else {
        v16bf b0 = load_frag(s_b0_ih, ihrs, lane, ki * 32);
        v16bf b1 = load_frag(s_b1_ih, ihrs, lane, ki * 32);
        acc0 = wmma_bf16(af[ki], b0, acc0);
        acc1 = wmma_bf16(af[ki], b1, acc1);
      }
    }
#pragma unroll
    for (int ki = 0; ki < KR; ++ki) {
      acc0 = wmma_bf16(af[KI + ki], whr0[ki], acc0);
      acc1 = wmma_bf16(af[KI + ki], whr1[ki], acc1);
    }

    // C/D layout: elem r of lane L is D[m = (L>>4)*8 + r][n = L&15]
    {
      const int r8 = (lane >> 4) << 3;
      const int nc = lane & 15;
      float* g0 = gates + (size_t)(mbase + r8) * kG4 + (ncol0 + nt0 * 16 + nc);
      float* g1 = gates + (size_t)(mbase + r8) * kG4 + (ncol0 + nt1 * 16 + nc);
#pragma unroll
      for (int r = 0; r < 8; ++r) {
        g0[(size_t)r * kG4] = acc0[r];
        g1[(size_t)r * kG4] = acc1[r];
      }
    }

    grid_barrier(bar);

    // ============ phase 2: cell update (c resident in registers) ===========
#pragma unroll
    for (int k = 0; k < CPT; ++k) {
      const int i = i0 + k * (NWG * NTHR);
      const int b = i / kH;
      const int j = i % kH;
      const float* gr = gates + (size_t)b * kG4;
      float iv = sigmoidf_(gr[j]           + bias[j]);
      float fv = sigmoidf_(gr[kH + j]      + bias[kH + j]);
      float gv = tanhf(gr[2 * kH + j]      + bias[2 * kH + j]);
      float ov = sigmoidf_(gr[3 * kH + j]  + bias[3 * kH + j]);
      float c  = fv * creg[k] + iv * gv;
      creg[k]  = c;
      float h  = ov * tanhf(c);
      hbp[i] = (bf16_t)h;
      if (hs_out)
        hs_out[(size_t)b * (kT * kH) + (size_t)t * kH + j] = (bf16_t)h;
      if (t == kT - 1) hfp[i] = h;
    }

    grid_barrier(bar);
  }
}

// ---------------------------------------------------------------------------
// Final projection: out[b] = dot(h_last[b,:], Wlin) + blin. One wave per row.
// ---------------------------------------------------------------------------
__global__ __launch_bounds__(NTHR) void final_linear_kernel(
    const float* __restrict__ hf, const float* __restrict__ wlin,
    const float* __restrict__ blin, float* __restrict__ out) {
  const int row  = blockIdx.x * (NTHR / 32) + (threadIdx.x >> 5);
  const int lane = threadIdx.x & 31;
  if (row >= kB) return;
  const float* hr = hf + (size_t)row * kH;
  float s = 0.0f;
  for (int k = lane; k < kH; k += 32) s += hr[k] * wlin[k];
#pragma unroll
  for (int off = 16; off > 0; off >>= 1) s += __shfl_down(s, off, 32);
  if (lane == 0) out[row] = s + blin[0];
}

// ---------------------------------------------------------------------------
// Host launcher
// ---------------------------------------------------------------------------
extern "C" void kernel_launch(void* const* d_in, const int* in_sizes, int n_in,
                              void* d_out, int out_size, void* d_ws,
                              size_t ws_size, hipStream_t stream) {
  (void)in_sizes; (void)n_in; (void)out_size; (void)ws_size;

  const float* x    = (const float*)d_in[0];
  const float* Wih0 = (const float*)d_in[1];
  const float* Whh0 = (const float*)d_in[2];
  const float* bih0 = (const float*)d_in[3];
  const float* bhh0 = (const float*)d_in[4];
  const float* Wih1 = (const float*)d_in[5];
  const float* Whh1 = (const float*)d_in[6];
  const float* bih1 = (const float*)d_in[7];
  const float* bhh1 = (const float*)d_in[8];
  const float* Wlin = (const float*)d_in[9];
  const float* blin = (const float*)d_in[10];

  char* ws = (char*)d_ws;
  size_t off = 0;
  auto take = [&](size_t bytes) -> char* {
    char* p = ws + off;
    off += (bytes + 255) & ~(size_t)255;
    return p;
  };

  unsigned* bar = (unsigned*)take(256);
  bf16_t* xbf   = (bf16_t*)take((size_t)kB * kT * kD * 2);   // 32 MB
  bf16_t* hs1   = (bf16_t*)take((size_t)kB * kT * kH * 2);   // 128 MB
  bf16_t* wih0  = (bf16_t*)take((size_t)kG4 * kD * 2);
  bf16_t* whh0  = (bf16_t*)take((size_t)kG4 * kH * 2);
  bf16_t* wih1  = (bf16_t*)take((size_t)kG4 * kH * 2);
  bf16_t* whh1  = (bf16_t*)take((size_t)kG4 * kH * 2);
  float* bias0  = (float*)take((size_t)kG4 * 4);
  float* bias1  = (float*)take((size_t)kG4 * 4);
  float* gates  = (float*)take((size_t)kB * kG4 * 4);        // 1 MB (L2-resident)
  float* hfp    = (float*)take((size_t)kB * kH * 4);
  bf16_t* hbp   = (bf16_t*)take((size_t)kB * kH * 2);

  // Precision prep: bf16 copies of x and the four weight matrices; fused biases.
  cvt_kernel<<<2048, NTHR, 0, stream>>>(xbf, x, kB * kT * kD);
  cvt_kernel<<<256, NTHR, 0, stream>>>(wih0, Wih0, kG4 * kD);
  cvt_kernel<<<256, NTHR, 0, stream>>>(whh0, Whh0, kG4 * kH);
  cvt_kernel<<<256, NTHR, 0, stream>>>(wih1, Wih1, kG4 * kH);
  cvt_kernel<<<256, NTHR, 0, stream>>>(whh1, Whh1, kG4 * kH);
  bias_kernel<<<4, NTHR, 0, stream>>>(bias0, bih0, bhh0, kG4);
  bias_kernel<<<4, NTHR, 0, stream>>>(bias1, bih1, bhh1, kG4);

  // Layer 0: din=64, ALL weights pinned in VGPRs (no LDS needed).
  zero_state_kernel<<<64, NTHR, 0, stream>>>(hbp, bar);
  lstm_layer_kernel<kD, true><<<NWG, NTHR, 0, stream>>>(
      xbf, wih0, whh0, bias0, gates, hfp, hbp, hs1, bar);

  // Layer 1: din=256, Whh in VGPRs, Wih slice in LDS (async fill once).
  const size_t shmem1 = (size_t)(64 * (kH + LDS_PAD)) * 2;  // 33 KB
  zero_state_kernel<<<64, NTHR, 0, stream>>>(hbp, bar);
  lstm_layer_kernel<kH, false><<<NWG, NTHR, shmem1, stream>>>(
      hs1, wih1, whh1, bias1, gates, hfp, hbp, nullptr, bar);

  // out[b] = h_T[b,:] . Wlin + blin
  final_linear_kernel<<<kB / 8, NTHR, 0, stream>>>(hfp, Wlin, blin,
                                                   (float*)d_out);
}
